// CRFParallel_15272903704594
// MI455X (gfx1250) — matrix operand order repaired and checked by reading, not compiled, not run on hardware
//
#include <hip/hip_runtime.h>

// ---------------------------------------------------------------------------
// BiLSTM-CRF decode for MI455X (gfx1250, wave32, WMMA + TDM).
//   V=100000, E=256, H=256, L=18, T=4096
// Pipeline:
//   1) convert weights fp32->bf16 (row-major)
//   2) gather x = emb[feats] -> bf16
//   3) pre = x @ w_ih^T + b  via v_wmma_f32_16x16x32_bf16 (fp32 accum)
//   4) sequential LSTM recurrence, 1 WGP per direction, matvec via WMMA;
//      pre[t] rows double-buffered into LDS with tensor_load_to_lds (TDM),
//      overlapped one step ahead and drained with s_wait_tensorcnt.
//   5) emit = hcat @ W_out^T + b_out   (tiny, VALU)
//   6) Viterbi forward + backtrace     (tiny, sequential, 1 wave)
// ---------------------------------------------------------------------------

typedef __attribute__((ext_vector_type(16))) __bf16 v16bf;
typedef __attribute__((ext_vector_type(8)))  float  v8f;
typedef unsigned int u32x4 __attribute__((ext_vector_type(4)));
typedef int          i32x8 __attribute__((ext_vector_type(8)));
typedef int          i32x4 __attribute__((ext_vector_type(4)));

union BF16x16 {
  v16bf v;
  uint4 q[2];
};

static constexpr int kT  = 4096;
static constexpr int kE  = 256;
static constexpr int kH  = 256;
static constexpr int kG  = 1024;   // 4*H gate width
static constexpr int kL  = 18;
static constexpr int kSTART = kL - 2;
static constexpr int kSTOP  = kL - 1;

__device__ __forceinline__ unsigned short bf16_rne(float f) {
  unsigned int u = __float_as_uint(f);
  u += 0x7FFFu + ((u >> 16) & 1u);
  return (unsigned short)(u >> 16);
}

// Issue a 1-D TDM load: kG floats from `src` -> LDS at byte offset `lds_off`.
// D# layout per CDNA5 ISA.
// This toolchain exposes the 6-arg builtin:
//   (uint32x4 g0, int32x8 g1, int32x4 g2, int32x4 g3, int32x8 pad, i32 cpol)
__device__ __forceinline__ void tdm_load_row(const float* src, unsigned lds_off) {
  const unsigned long long ga = (unsigned long long)(uintptr_t)src;
  u32x4 g0;
  g0[0] = 1u;                                   // count=1 (valid descriptor)
  g0[1] = lds_off;                              // lds_addr (bytes)
  g0[2] = (unsigned)ga;                         // global_addr[31:0]
  g0[3] = ((unsigned)(ga >> 32) & 0x01FFFFFFu)  // global_addr[56:32]
          | 0x80000000u;                        // type=2 ("image")
  i32x8 g1;
  g1[0] = 0x00020000;            // workgroup_mask=0, data_size=2 (4 bytes)
  g1[1] = (int)((kG & 0xFFFF) << 16);           // tensor_dim0[15:0] @ bits 63:48
  g1[2] = (int)(((kG >> 16) & 0xFFFF)           // tensor_dim0[31:16]
                | (1u << 16));                  // tensor_dim1 = 1
  g1[3] = (int)((unsigned)kG << 16);            // tile_dim0 = 1024 @ bits 127:112
  g1[4] = 0;                                    // tile_dim1=0, tile_dim2=0
  g1[5] = kG;                                   // tensor_dim0_stride[31:0]
  g1[6] = 0;
  g1[7] = 0;
  const i32x4 z4 = {};
  const i32x8 z8 = {};
  __builtin_amdgcn_tensor_load_to_lds(g0, g1, z4, z4, z8, 0);
}

// ---------------------------------------------------------------------------
// fp32 -> bf16 conversion
// ---------------------------------------------------------------------------
__global__ void cvt_bf16_kernel(const float* __restrict__ src,
                                unsigned short* __restrict__ dst, int n) {
  int i = blockIdx.x * blockDim.x + threadIdx.x;
  if (i < n) dst[i] = bf16_rne(src[i]);
}

// ---------------------------------------------------------------------------
// x = emb[feats]  (gather + convert to bf16, row-major T x E)
// ---------------------------------------------------------------------------
__global__ void gather_embed_kernel(const int* __restrict__ feats,
                                    const float* __restrict__ emb,
                                    unsigned short* __restrict__ xb) {
  int t = blockIdx.x;
  int e = threadIdx.x;
  int v = feats[t];
  xb[t * kE + e] = bf16_rne(emb[(size_t)v * kE + e]);
}

// ---------------------------------------------------------------------------
// pre = x @ w_ih^T + b : M=4096, N=1024, K=256, both directions.
// One wave per 16x16 output tile; K consumed in 8 chunks of 32 via WMMA bf16.
// grid = (Mtiles=256, Ntiles/4=16, dir=2), block = 128 (4 waves).
// ---------------------------------------------------------------------------
__global__ __launch_bounds__(128) void input_gemm_kernel(
    const unsigned short* __restrict__ xb,
    const unsigned short* __restrict__ wih_f,
    const unsigned short* __restrict__ wih_b,
    const float* __restrict__ bias_f, const float* __restrict__ bias_b,
    float* __restrict__ pre_f, float* __restrict__ pre_b) {
  const int wave = threadIdx.x >> 5;
  const int lane = threadIdx.x & 31;
  const int g    = lane >> 4;       // lane half-group
  const int col  = lane & 15;
  const int m0   = blockIdx.x * 16;
  const int n0   = (blockIdx.y * 4 + wave) * 16;

  const unsigned short* W = blockIdx.z ? wih_b : wih_f;
  const float* bias       = blockIdx.z ? bias_b : bias_f;
  float* pre              = blockIdx.z ? pre_b : pre_f;

  v8f acc = {};
#pragma unroll
  for (int c = 0; c < 8; ++c) {
    const int k0 = c * 32;
    BF16x16 A, B;
    // A (16x32 bf16): lane row = col; K = 16g..16g+15 (contiguous 16 elems)
    const unsigned short* ap = xb + (size_t)(m0 + col) * kE + k0;
    A.q[0] = *(const uint4*)(ap + 16 * g);
    A.q[1] = *(const uint4*)(ap + 16 * g + 8);
    // B (32x16 bf16): lane column = col; K pairs 2v + 16g (contiguous 16 elems)
    const unsigned short* bp = W + (size_t)(n0 + col) * kE + k0 + 16 * g;
    B.q[0] = *(const uint4*)(bp);
    B.q[1] = *(const uint4*)(bp + 8);
    acc = __builtin_amdgcn_wmma_f32_16x16x32_bf16(false, A.v, false, B.v,
                                                  (short)0, acc, false, false);
  }
  const float bv = bias[n0 + col];
#pragma unroll
  for (int r = 0; r < 8; ++r) {
    // C layout: VGPR r -> (M = r + 8g, N = col)
    const int m = m0 + r + 8 * g;
    pre[(size_t)m * kG + n0 + col] = acc[r] + bv;
  }
}

// ---------------------------------------------------------------------------
// Sequential LSTM recurrence. grid = 2 blocks (fwd/bwd), block = 1024 (32 waves).
// Per step: g = h @ w_hh^T via WMMA (A = h replicated into 16 rows, broadcast
// from LDS); wave w owns gate columns [32w, 32w+32). pre rows double-buffered
// in LDS via TDM (issued one step ahead by wave 0). Elementwise by threads
// 0..255; h re-broadcast through LDS as bf16; c stays in registers.
// ---------------------------------------------------------------------------
__global__ __launch_bounds__(1024) void bilstm_recur_kernel(
    const float* __restrict__ pre_f, const float* __restrict__ pre_b,
    const unsigned short* __restrict__ whh_f,
    const unsigned short* __restrict__ whh_b,
    float* __restrict__ hcat) {
  const bool rev = (blockIdx.x != 0);
  const float* pre          = rev ? pre_b : pre_f;
  const unsigned short* W   = rev ? whh_b : whh_f;

  const int tid  = threadIdx.x;
  const int wave = tid >> 5;
  const int lane = tid & 31;
  const int g    = lane >> 4;
  const int col  = lane & 15;

  __shared__ __align__(16) unsigned short hbf[kH];  // h broadcast (bf16)
  __shared__ float gsh[kG];                         // gate pre-activations
  __shared__ __align__(16) float preLds[2][kG];     // TDM double buffer

  // Per-wave B operand base pointers (two 16-col tiles: n = 32w .. 32w+31)
  const unsigned short* w0 = W + (size_t)(32 * wave + col) * kH;
  const unsigned short* w1 = W + (size_t)(32 * wave + 16 + col) * kH;

  float c_state = 0.0f;
  if (tid < kH) hbf[tid] = 0;
  if (wave == 0) {
    // Stage pre row for step 0 into buffer 0.
    const int t0 = rev ? (kT - 1) : 0;
    tdm_load_row(pre + (size_t)t0 * kG, (unsigned)(uintptr_t)&preLds[0][0]);
    __builtin_amdgcn_s_wait_tensorcnt(0);
  }
  __syncthreads();

  for (int s = 0; s < kT; ++s) {
    const int t = rev ? (kT - 1 - s) : s;

    v8f acc0 = {};
    v8f acc1 = {};
#pragma unroll
    for (int c = 0; c < 8; ++c) {
      const int k0 = c * 32;
      BF16x16 A, B0, B1;
      // A = h[k0..k0+31] replicated across all 16 rows (LDS broadcast reads)
      A.q[0] = *(const uint4*)(hbf + k0 + 16 * g);
      A.q[1] = *(const uint4*)(hbf + k0 + 16 * g + 8);
      // B tiles streamed from L2 (w_hh bf16 = 512 KB, fully L2-resident)
      B0.q[0] = *(const uint4*)(w0 + k0 + 16 * g);
      B0.q[1] = *(const uint4*)(w0 + k0 + 16 * g + 8);
      B1.q[0] = *(const uint4*)(w1 + k0 + 16 * g);
      B1.q[1] = *(const uint4*)(w1 + k0 + 16 * g + 8);
      acc0 = __builtin_amdgcn_wmma_f32_16x16x32_bf16(false, A.v, false, B0.v,
                                                     (short)0, acc0, false, false);
      acc1 = __builtin_amdgcn_wmma_f32_16x16x32_bf16(false, A.v, false, B1.v,
                                                     (short)0, acc1, false, false);
    }
    // Kick off the DMA for the next step's pre row (overlaps this step).
    if (wave == 0 && s + 1 < kT) {
      const int tn = rev ? (kT - 2 - s) : (s + 1);
      tdm_load_row(pre + (size_t)tn * kG,
                   (unsigned)(uintptr_t)&preLds[(s + 1) & 1][0]);
    }
    // All 16 rows of D are identical; row value for column n sits in C VGPR0.
    if (lane < 16) gsh[32 * wave + col]      = acc0[0];
    else           gsh[32 * wave + 16 + col] = acc1[0];
    __syncthreads();

    if (tid < kH) {
      const float* prow = &preLds[s & 1][0];
      const float gi = gsh[tid]            + prow[tid];
      const float gf = gsh[kH + tid]       + prow[kH + tid];
      const float gc = gsh[2 * kH + tid]   + prow[2 * kH + tid];
      const float go = gsh[3 * kH + tid]   + prow[3 * kH + tid];
      const float i_g = 1.0f / (1.0f + __expf(-gi));
      const float f_g = 1.0f / (1.0f + __expf(-gf));
      const float o_g = 1.0f / (1.0f + __expf(-go));
      c_state = f_g * c_state + i_g * tanhf(gc);
      const float h = o_g * tanhf(c_state);
      hbf[tid] = bf16_rne(h);
      hcat[(size_t)t * (2 * kH) + (rev ? kH : 0) + tid] = h;
    }
    // Drain the next-step DMA before the barrier so every wave may read it.
    if (wave == 0) __builtin_amdgcn_s_wait_tensorcnt(0);
    __syncthreads();
  }
}

// ---------------------------------------------------------------------------
// emit[t][l] = hcat[t] . W_out[l] + b_out[l]   (T x 18, stride 32 in ws)
// ---------------------------------------------------------------------------
__global__ void emit_kernel(const float* __restrict__ hcat,
                            const float* __restrict__ Wout,
                            const float* __restrict__ bout,
                            float* __restrict__ emit) {
  const int t = blockIdx.x;
  const int l = threadIdx.x;
  if (l < kL) {
    const float* hrow = hcat + (size_t)t * (2 * kH);
    const float* wrow = Wout + (size_t)l * (2 * kH);
    float s = bout[l];
#pragma unroll 8
    for (int k = 0; k < 2 * kH; ++k) s += hrow[k] * wrow[k];
    emit[t * 32 + l] = s;
  }
}

// ---------------------------------------------------------------------------
// Viterbi forward + backtrace. 1 block, 1 wave (lanes 0..17 active).
// out[0] = path_score, out[1..T] = best_path (as float).
// ---------------------------------------------------------------------------
__global__ void viterbi_kernel(const float* __restrict__ emit,
                               const float* __restrict__ trans,
                               unsigned char* __restrict__ bp,
                               float* __restrict__ out) {
  __shared__ float tr[kL * kL];
  __shared__ float fv[kL];
  const int tid = threadIdx.x;
  for (int i = tid; i < kL * kL; i += 32) tr[i] = trans[i];
  if (tid < kL) fv[tid] = (tid == kSTART) ? 0.0f : -10000.0f;
  __syncthreads();

  for (int t = 0; t < kT; ++t) {
    float best = -3.0e38f;
    int bi = 0;
    float nv = 0.0f;
    if (tid < kL) {
      for (int l = 0; l < kL; ++l) {
        const float s = fv[l] + tr[l * kL + tid];
        if (s > best) { best = s; bi = l; }
      }
      nv = best + emit[t * 32 + tid];
      bp[t * kL + tid] = (unsigned char)bi;
    }
    __syncthreads();
    if (tid < kL) fv[tid] = nv;
    __syncthreads();
  }

  if (tid == 0) {
    float best = -3.0e38f;
    int bt = 0;
    for (int l = 0; l < kL; ++l) {
      const float s = fv[l] + tr[l * kL + kSTOP];
      if (s > best) { best = s; bt = l; }
    }
    out[0] = best;
    int tag = bt;
    for (int t = kT - 1; t >= 0; --t) {
      out[1 + t] = (float)tag;
      tag = bp[t * kL + tag];
    }
  }
}

// ---------------------------------------------------------------------------
// Launcher
// ---------------------------------------------------------------------------
extern "C" void kernel_launch(void* const* d_in, const int* in_sizes, int n_in,
                              void* d_out, int out_size, void* d_ws, size_t ws_size,
                              hipStream_t stream) {
  const int*   feats  = (const int*)d_in[0];
  const float* emb    = (const float*)d_in[1];
  const float* w_ih_f = (const float*)d_in[2];
  const float* w_hh_f = (const float*)d_in[3];
  const float* b_f    = (const float*)d_in[4];
  const float* w_ih_b = (const float*)d_in[5];
  const float* w_hh_b = (const float*)d_in[6];
  const float* b_b    = (const float*)d_in[7];
  const float* W_out  = (const float*)d_in[8];
  const float* b_out  = (const float*)d_in[9];
  const float* trans  = (const float*)d_in[10];
  float* out = (float*)d_out;

  char* ws = (char*)d_ws;
  size_t off = 0;
  auto alloc = [&](size_t bytes) -> void* {
    void* p = ws + off;
    off = (off + bytes + 255) & ~(size_t)255;
    return p;
  };

  unsigned short* xb     = (unsigned short*)alloc((size_t)kT * kE * 2);
  unsigned short* wihf16 = (unsigned short*)alloc((size_t)kG * kE * 2);
  unsigned short* wihb16 = (unsigned short*)alloc((size_t)kG * kE * 2);
  unsigned short* whhf16 = (unsigned short*)alloc((size_t)kG * kH * 2);
  unsigned short* whhb16 = (unsigned short*)alloc((size_t)kG * kH * 2);
  float* pre_f           = (float*)alloc((size_t)kT * kG * 4);
  float* pre_b           = (float*)alloc((size_t)kT * kG * 4);
  float* hcat            = (float*)alloc((size_t)kT * 2 * kH * 4);
  float* emit            = (float*)alloc((size_t)kT * 32 * 4);
  unsigned char* bp      = (unsigned char*)alloc((size_t)kT * kL);

  const int wn = kG * kE;  // 262144 elements per weight matrix
  cvt_bf16_kernel<<<(wn + 255) / 256, 256, 0, stream>>>(w_ih_f, wihf16, wn);
  cvt_bf16_kernel<<<(wn + 255) / 256, 256, 0, stream>>>(w_ih_b, wihb16, wn);
  cvt_bf16_kernel<<<(wn + 255) / 256, 256, 0, stream>>>(w_hh_f, whhf16, wn);
  cvt_bf16_kernel<<<(wn + 255) / 256, 256, 0, stream>>>(w_hh_b, whhb16, wn);

  gather_embed_kernel<<<kT, kE, 0, stream>>>(feats, emb, xb);

  input_gemm_kernel<<<dim3(kT / 16, (kG / 16) / 4, 2), 128, 0, stream>>>(
      xb, wihf16, wihb16, b_f, b_b, pre_f, pre_b);

  bilstm_recur_kernel<<<2, 1024, 0, stream>>>(pre_f, pre_b, whhf16, whhb16, hcat);

  emit_kernel<<<kT, 32, 0, stream>>>(hcat, W_out, b_out, emit);

  viterbi_kernel<<<1, 32, 0, stream>>>(emit, trans, bp, out);
}
